// DebedderNeuronGroup_index_45981919871513
// MI455X (gfx1250) — matrix operand order repaired and checked by reading, not compiled, not run on hardware
//
#include <hip/hip_runtime.h>

typedef __attribute__((ext_vector_type(16))) __bf16 v16bf;
typedef __attribute__((ext_vector_type(8)))  float  v8f;

#define I_DIM_TOTAL 1422218ull
#define DM   512   // d_model (K dimension)
#define NTOK 714   // total kernels (token axis of x)

// ---------------------------------------------------------------------------
// helpers
// ---------------------------------------------------------------------------
__device__ __forceinline__ float4 ld4(const float* p) { return *(const float4*)p; }

__device__ __forceinline__ v16bf pack16(float4 a, float4 b, float4 c, float4 d) {
  v16bf v;
  v[0]=(__bf16)a.x;  v[1]=(__bf16)a.y;  v[2]=(__bf16)a.z;  v[3]=(__bf16)a.w;
  v[4]=(__bf16)b.x;  v[5]=(__bf16)b.y;  v[6]=(__bf16)b.z;  v[7]=(__bf16)b.w;
  v[8]=(__bf16)c.x;  v[9]=(__bf16)c.y;  v[10]=(__bf16)c.z; v[11]=(__bf16)c.w;
  v[12]=(__bf16)d.x; v[13]=(__bf16)d.y; v[14]=(__bf16)d.z; v[15]=(__bf16)d.w;
  return v;
}

// A fragment, 16-bit, 16x32 (MxK): lane holds K = {ah..ah+7} and {16+ah..16+ah+7}
// (p already offset by kk + ah).  bf16 source: two 16B loads.
__device__ __forceinline__ v16bf load_a_bf16(const __bf16* p) {
  union { v16bf v; uint4 q[2]; } u;
  u.q[0] = *(const uint4*)(p);
  u.q[1] = *(const uint4*)(p + 16);
  return u.v;
}
// f32 source fallback: four b128 loads + cvt to bf16 in-register
__device__ __forceinline__ v16bf load_a_f32(const float* p) {
  return pack16(ld4(p), ld4(p + 4), ld4(p + 16), ld4(p + 20));
}

// B fragment, 16-bit, 32x16 (KxN): lane = column N, holds K = {bh..bh+7} and
// {16+bh..16+bh+7} (bh = 8*(lane>=16)), same order as A.
__device__ __forceinline__ v16bf load_b_bf16(const __bf16* p) {
  union { v16bf v; uint4 q[2]; } u;
  u.q[0] = *(const uint4*)(p);
  u.q[1] = *(const uint4*)(p + 16);
  return u.v;
}
__device__ __forceinline__ v16bf load_b_f32(const float* p) {
  return pack16(ld4(p), ld4(p + 4), ld4(p + 16), ld4(p + 20));
}

// ---------------------------------------------------------------------------
// f32 -> bf16 bulk conversion (vectorized: 8 elems / thread, 16B stores)
// ---------------------------------------------------------------------------
__global__ void cvt_f32_bf16_kernel(const float* __restrict__ src,
                                    unsigned short* __restrict__ dstu,
                                    long long n) {
  long long i = ((long long)blockIdx.x * blockDim.x + threadIdx.x) * 8;
  if (i + 8 <= n) {
    float4 a = *(const float4*)(src + i);
    float4 b = *(const float4*)(src + i + 4);
    union { __bf16 h[8]; uint4 q; } u;
    u.h[0]=(__bf16)a.x; u.h[1]=(__bf16)a.y; u.h[2]=(__bf16)a.z; u.h[3]=(__bf16)a.w;
    u.h[4]=(__bf16)b.x; u.h[5]=(__bf16)b.y; u.h[6]=(__bf16)b.z; u.h[7]=(__bf16)b.w;
    *(uint4*)(dstu + i) = u.q;
  } else if (i < n) {
    __bf16* dst = (__bf16*)dstu;
    for (; i < n; ++i) dst[i] = (__bf16)src[i];
  }
}

// ---------------------------------------------------------------------------
// Per-layer GEMM:  out[b,k,o] = sum_d x[b,off+k,d] * W[o,d] + bias[o]
// scattered into y.  One wave = 32(M) x 64(N) macro-tile (2x4 WMMA C tiles),
// K = 512 in 16 steps of 32 via v_wmma_f32_16x16x32_bf16.
// Block = 256 threads = 8 waves = 256 M rows; M_l divisible by 256 for all layers.
// ---------------------------------------------------------------------------
template<bool SRC_BF16>
__global__ __launch_bounds__(256) void layer_gemm_wmma(
    const void* __restrict__ xin,   // bf16 (fast) or f32 (fallback), [*, NTOK or kn, DM]
    const void* __restrict__ win,   // bf16 or f32, [i_dim, DM]
    const float* __restrict__ bias, // f32 [i_dim]
    float* __restrict__ y,          // f32 [128, I_DIM_TOTAL]
    int off, int kn, int i_dim, int ksci, int sbase)
{
  const int lane = threadIdx.x & 31;
  const int wv   = threadIdx.x >> 5;
  const int t0   = blockIdx.x * 256 + wv * 32;  // first token row of this wave
  const int n0   = blockIdx.y * 64;             // first output column of this N group
  const int lm   = lane & 15;
  const int hi   = lane >> 4;                   // 0 / 1
  const int ah   = hi * 8;                      // A-fragment K-half offset
  const int bh   = hi * 8;                      // B-fragment K-half offset

  // ---- A row addresses (2 M tiles); token row -> (batch, k) -> flat x row
  const unsigned ukn = (unsigned)kn;
  unsigned rA0 = (unsigned)(t0 + lm);
  unsigned rA1 = rA0 + 16u;
  unsigned bA0 = rA0 / ukn, bA1 = rA1 / ukn;
  size_t gA0 = ((size_t)bA0 * NTOK + (size_t)off + (rA0 - bA0 * ukn)) * DM + (size_t)ah;
  size_t gA1 = ((size_t)bA1 * NTOK + (size_t)off + (rA1 - bA1 * ukn)) * DM + (size_t)ah;

  // ---- B row addresses (4 N tiles), clamped; OOB columns never stored
  int oc_[4]; size_t gB[4];
#pragma unroll
  for (int n = 0; n < 4; ++n) {
    int o = n0 + n * 16 + lm;
    oc_[n] = o;
    int oc = (o < i_dim) ? o : 0;
    gB[n] = (size_t)oc * DM + (size_t)bh;
  }

  const v8f vz = {0.f, 0.f, 0.f, 0.f, 0.f, 0.f, 0.f, 0.f};
  v8f acc[2][4];
#pragma unroll
  for (int m = 0; m < 2; ++m)
#pragma unroll
    for (int n = 0; n < 4; ++n) acc[m][n] = vz;

  // ---- main K loop: 16 steps, 8 WMMAs per step
  if constexpr (SRC_BF16) {
    const __bf16* X  = (const __bf16*)xin;
    const __bf16* Wm = (const __bf16*)win;
#pragma unroll 4
    for (int kk = 0; kk < DM; kk += 32) {
      v16bf a0 = load_a_bf16(X + gA0 + kk);
      v16bf a1 = load_a_bf16(X + gA1 + kk);
#pragma unroll
      for (int n = 0; n < 4; ++n) {
        v16bf bm = load_b_bf16(Wm + gB[n] + kk);
        acc[0][n] = __builtin_amdgcn_wmma_f32_16x16x32_bf16(
            false, a0, false, bm, (short)0, acc[0][n], false, false);
        acc[1][n] = __builtin_amdgcn_wmma_f32_16x16x32_bf16(
            false, a1, false, bm, (short)0, acc[1][n], false, false);
      }
    }
  } else {
    const float* X  = (const float*)xin;
    const float* Wm = (const float*)win;
#pragma unroll 2
    for (int kk = 0; kk < DM; kk += 32) {
      v16bf a0 = load_a_f32(X + gA0 + kk);
      v16bf a1 = load_a_f32(X + gA1 + kk);
#pragma unroll
      for (int n = 0; n < 4; ++n) {
        v16bf bm = load_b_f32(Wm + gB[n] + kk);
        acc[0][n] = __builtin_amdgcn_wmma_f32_16x16x32_bf16(
            false, a0, false, bm, (short)0, acc[0][n], false, false);
        acc[1][n] = __builtin_amdgcn_wmma_f32_16x16x32_bf16(
            false, a1, false, bm, (short)0, acc[1][n], false, false);
      }
    }
  }

  // ---- scatter-store with bias add (closed-form column mapping).
  // C layout: VGPR r of lane -> M = r + 8*hi (in tile), N = lane%16.
  const size_t knksci = (size_t)kn * (size_t)ksci;
#pragma unroll
  for (int m = 0; m < 2; ++m) {
    unsigned rb = (unsigned)(t0 + m * 16 + ah);   // first M row of this lane's 8 values
    unsigned bb = rb / ukn;
    int k0 = (int)(rb - bb * ukn);
#pragma unroll
    for (int n = 0; n < 4; ++n) {
      int o = oc_[n];
      if (o >= i_dim) continue;
      float bv = bias[o];
      bool isw = (o < ksci);
#pragma unroll
      for (int r = 0; r < 8; ++r) {
        int k = k0 + r;
        unsigned bi = bb;
        if (k >= kn) { k -= kn; bi += 1u; }     // wraps at most once within 8 rows
        size_t col = isw ? ((size_t)sbase + (size_t)k * (size_t)ksci + (size_t)o)
                         : ((size_t)sbase + knksci + (size_t)k);
        __builtin_nontemporal_store(acc[m][n][r] + bv,
                                    y + (size_t)bi * I_DIM_TOTAL + col);
      }
    }
  }
}

// ---------------------------------------------------------------------------
// launch
// ---------------------------------------------------------------------------
extern "C" void kernel_launch(void* const* d_in, const int* in_sizes, int n_in,
                              void* d_out, int out_size, void* d_ws, size_t ws_size,
                              hipStream_t stream) {
  (void)in_sizes; (void)n_in; (void)out_size;

  const float* x = (const float*)d_in[0];
  const float* Wl[5]; const float* bl[5];
  for (int l = 0; l < 5; ++l) { Wl[l] = (const float*)d_in[1 + 2 * l];
                                bl[l] = (const float*)d_in[2 + 2 * l]; }
  float* y = (float*)d_out;

  static const int kn[5]   = {64, 128, 256, 256, 10};
  static const int idim[5] = {28, 577, 1153, 4097, 257};
  static const int ksci[5] = {27, 576, 1152, 4096, 256};
  static const int sb[5]   = {0, 1792, 75648, 370816, 1419648};
  static const int off[5]  = {0, 64, 192, 448, 704};

  const long long xe = 128LL * NTOK * DM;   // 46,792,704 elements
  long long we[5]; long long wtot = 0;
  for (int l = 0; l < 5; ++l) { we[l] = wtot; wtot += (long long)idim[l] * DM; }
  const size_t need = (size_t)(xe + wtot) * sizeof(unsigned short);  // ~95 MB

  const bool fast = (ws_size >= need);

  if (fast) {
    unsigned short* xb = (unsigned short*)d_ws;
    unsigned short* wb = xb + xe;
    // convert x and all W to bf16 once (streams through HBM once, ~12 us)
    {
      long long n = xe;
      int blocks = (int)((n / 8 + 255) / 256);
      cvt_f32_bf16_kernel<<<blocks, 256, 0, stream>>>(x, xb, n);
    }
    for (int l = 0; l < 5; ++l) {
      long long n = (long long)idim[l] * DM;
      int blocks = (int)((n / 8 + 255) / 256);
      cvt_f32_bf16_kernel<<<blocks, 256, 0, stream>>>(Wl[l], wb + we[l], n);
    }
    for (int l = 0; l < 5; ++l) {
      int M = 128 * kn[l];
      dim3 grid(M / 256, (idim[l] + 63) / 64, 1);
      layer_gemm_wmma<true><<<grid, 256, 0, stream>>>(
          (const void*)xb, (const void*)(wb + we[l]), bl[l], y,
          off[l], kn[l], idim[l], ksci[l], sb[l]);
    }
  } else {
    // scratch too small: convert f32 -> bf16 in-register inside the GEMM
    for (int l = 0; l < 5; ++l) {
      int M = 128 * kn[l];
      dim3 grid(M / 256, (idim[l] + 63) / 64, 1);
      layer_gemm_wmma<false><<<grid, 256, 0, stream>>>(
          (const void*)x, (const void*)Wl[l], bl[l], y,
          off[l], kn[l], idim[l], ksci[l], sb[l]);
    }
  }
}
